// ResGraphSAGE_2516850835980
// MI455X (gfx1250) — matrix operand order repaired and checked by reading, not compiled, not run on hardware
//
#include <hip/hip_runtime.h>

#define HID     128
#define IN_DIM  256
#define N_LAYERS 4
#define BN_EPS  1e-5f

typedef __attribute__((ext_vector_type(16))) _Float16 v16h;
typedef __attribute__((ext_vector_type(8)))  _Float16 v8h;
typedef __attribute__((ext_vector_type(8)))  float    v8f;

// LDS A-tile row stride: 264 halves = 528 B = 132 dwords -> bank stride 4.
// Lanes m=0..15 read rows m => banks 4m..4m+3 per ds_load_b128: conflict-free.
#define A_STRIDE 264

__device__ __forceinline__ v16h combine16(v8h lo, v8h hi) {
  v16h r;
#pragma unroll
  for (int i = 0; i < 8; ++i) { r[i] = lo[i]; r[i + 8] = hi[i]; }
  return r;
}

// A 16xK f16 fragment from LDS tile (CDNA5 §7.12.2):
//  elems 0..7  <-> K = k0 + g*8 + i
//  elems 8..15 <-> K = k0 + 16 + g*8 + i
__device__ __forceinline__ v16h load_a_frag_lds(const _Float16* Arow, int k0, int g) {
  v8h lo = *(const v8h*)(Arow + k0 + g * 8);
  v8h hi = *(const v8h*)(Arow + k0 + 16 + g * 8);
  return combine16(lo, hi);
}

// B 32x16 f16 fragment from pre-transposed f16 weights Wt[col][k]:
//  elems 0..7 <-> K = k0 + g*8 + i, elems 8..15 <-> K = k0 + 16 + g*8 + i -> 2x b128
__device__ __forceinline__ v16h load_b_frag(const _Float16* __restrict__ Wt,
                                            int col, int k0, int g, int ldk) {
  const _Float16* p = Wt + (size_t)col * ldk + k0 + g * 8;
  return combine16(*(const v8h*)p, *(const v8h*)(p + 16));
}

// ---------------------------------------------------------------------------
__global__ void fill_zero_kernel(float* __restrict__ p, size_t count) {
  size_t i = (size_t)blockIdx.x * blockDim.x + threadIdx.x;
  size_t stride = (size_t)gridDim.x * blockDim.x;
  for (; i < count; i += stride) p[i] = 0.0f;
}

// ---------------------------------------------------------------------------
// Pre-convert weights to f16, transposed to [col][k] (done once per launch).
//  Wti[c][k] = W_in[k][c]                         (c<128, k<256)
//  Wtl[l][c][k] = k<128 ? W_l[l][k][c] : W_r[l][k-128][c]
// ---------------------------------------------------------------------------
__global__ void convert_weights_kernel(const float* __restrict__ Wi,
                                       const float* __restrict__ Wl,
                                       const float* __restrict__ Wr,
                                       _Float16* __restrict__ Wti,
                                       _Float16* __restrict__ Wtl) {
  int idx = blockIdx.x * blockDim.x + threadIdx.x;
  const int total_in = HID * IN_DIM;  // 32768
  if (idx < total_in) {
    int c = idx >> 8, k = idx & 255;
    Wti[(size_t)c * IN_DIM + k] = (_Float16)Wi[(size_t)k * HID + c];
  }
  int idx2 = idx - total_in;
  if (idx2 >= 0 && idx2 < N_LAYERS * HID * 256) {
    int l = idx2 >> 15;        // 128*256 = 1<<15
    int rem = idx2 & 32767;
    int c = rem >> 8, k = rem & 255;
    float v = (k < HID) ? Wl[((size_t)l * HID + k) * HID + c]
                        : Wr[((size_t)l * HID + (k - HID)) * HID + c];
    Wtl[((size_t)l * HID + c) * 256 + k] = (_Float16)v;
  }
}

// ---------------------------------------------------------------------------
__global__ void count_kernel(const int* __restrict__ dst, float* __restrict__ cnt, int E) {
  int e = blockIdx.x * blockDim.x + threadIdx.x;
  if (e < E) atomicAdd(&cnt[dst[e]], 1.0f);
}

// ---------------------------------------------------------------------------
// Edge scatter-add: 32 lanes/edge, float4 gather (L2-resident h), f32 atomics.
// global_prefetch for a source row ~64 edges ahead.
// ---------------------------------------------------------------------------
__global__ __launch_bounds__(256) void scatter_kernel(
    const float* __restrict__ h, const int* __restrict__ src,
    const int* __restrict__ dst, float* __restrict__ agg, int E) {
  int t = blockIdx.x * blockDim.x + threadIdx.x;
  int e = t >> 5;
  if (e >= E) return;
  int lane = t & 31;
  int s = src[e];
  int d = dst[e];
  if (lane == 0 && e + 64 < E) {
    __builtin_prefetch(h + (size_t)src[e + 64] * HID, 0, 3);
  }
  const float4* hs = (const float4*)(h + (size_t)s * HID);
  float4 v = hs[lane];
  float* ad = agg + (size_t)d * HID + lane * 4;
  atomicAdd(ad + 0, v.x);
  atomicAdd(ad + 1, v.y);
  atomicAdd(ad + 2, v.z);
  atomicAdd(ad + 3, v.w);
}

// ---------------------------------------------------------------------------
// Stage one row's 16-float chunk (64 B contiguous per thread) into the LDS
// A-tile as f16, with optional scale.
// ---------------------------------------------------------------------------
__device__ __forceinline__ void stage16(const float* __restrict__ srcp, float scale,
                                        _Float16* dstp) {
  const float4* s4 = (const float4*)srcp;
  float4 f0 = s4[0], f1 = s4[1], f2 = s4[2], f3 = s4[3];
  v8h h0, h1;
  h0[0] = (_Float16)(f0.x * scale); h0[1] = (_Float16)(f0.y * scale);
  h0[2] = (_Float16)(f0.z * scale); h0[3] = (_Float16)(f0.w * scale);
  h0[4] = (_Float16)(f1.x * scale); h0[5] = (_Float16)(f1.y * scale);
  h0[6] = (_Float16)(f1.z * scale); h0[7] = (_Float16)(f1.w * scale);
  h1[0] = (_Float16)(f2.x * scale); h1[1] = (_Float16)(f2.y * scale);
  h1[2] = (_Float16)(f2.z * scale); h1[3] = (_Float16)(f2.w * scale);
  h1[4] = (_Float16)(f3.x * scale); h1[5] = (_Float16)(f3.y * scale);
  h1[6] = (_Float16)(f3.z * scale); h1[7] = (_Float16)(f3.w * scale);
  *(v8h*)(dstp) = h0;
  *(v8h*)(dstp + 8) = h1;
}

// ---------------------------------------------------------------------------
// Input layer: h = relu(x @ W_in + b_in), x:[n,256] f32, Wti:[128][256] f16.
// Block = 16 rows; 8 waves = 8 column tiles; K=256 -> 8 WMMA steps.
// ---------------------------------------------------------------------------
__global__ __launch_bounds__(256) void in_gemm_kernel(
    const float* __restrict__ x, const _Float16* __restrict__ Wti,
    const float* __restrict__ bi, float* __restrict__ h, int n) {
  __shared__ __align__(16) _Float16 As[16][A_STRIDE];

  // cooperative stage: thread t -> row t/16, k chunk (t%16)*16
  {
    int r = threadIdx.x >> 4;
    int kc = (threadIdx.x & 15) * 16;
    int row = blockIdx.x * 16 + r;
    if (row >= n) row = n - 1;
    stage16(x + (size_t)row * IN_DIM + kc, 1.0f, &As[r][kc]);
  }
  __syncthreads();

  const int wave = threadIdx.x >> 5;
  const int lane = threadIdx.x & 31;
  const int g = lane >> 4;
  const int m = lane & 15;
  const int col = wave * 16 + m;

  v8f c = {};
#pragma unroll
  for (int k0 = 0; k0 < IN_DIM; k0 += 32) {
    v16h a = load_a_frag_lds(&As[m][0], k0, g);
    v16h b = load_b_frag(Wti, col, k0, g, IN_DIM);
    c = __builtin_amdgcn_wmma_f32_16x16x32_f16(false, a, false, b, (short)0, c,
                                               false, false);
  }
  const float bias = bi[col];
  const int rb = blockIdx.x * 16 + g * 8;
#pragma unroll
  for (int r = 0; r < 8; ++r) {
    int orow = rb + r;
    if (orow < n) h[(size_t)orow * HID + col] = fmaxf(c[r] + bias, 0.0f);
  }
}

// ---------------------------------------------------------------------------
// Fused SAGE layer:
//   hout = relu(BN( (agg/cnt)@W_l + b_l + h@W_r + b_r )) + h
// LDS A-tile = [mean | h] (K=256), Wt = [W_l ; W_r] pre-transposed f16.
// ---------------------------------------------------------------------------
__global__ __launch_bounds__(256) void layer_kernel(
    const float* __restrict__ h, const float* __restrict__ agg,
    const float* __restrict__ cnt, const _Float16* __restrict__ Wt,
    const float* __restrict__ bl, const float* __restrict__ br,
    const float* __restrict__ gamma, const float* __restrict__ beta,
    const float* __restrict__ mu, const float* __restrict__ var,
    float* __restrict__ hout, int n) {
  __shared__ __align__(16) _Float16 As[16][A_STRIDE];

  {
    int r = threadIdx.x >> 4;
    int kc = (threadIdx.x & 15) * 16;
    int row = blockIdx.x * 16 + r;
    if (row >= n) row = n - 1;
    if (kc < HID) {  // mean part: agg * (1/max(cnt,1))
      float inv = 1.0f / fmaxf(cnt[row], 1.0f);
      stage16(agg + (size_t)row * HID + kc, inv, &As[r][kc]);
    } else {         // self part: h
      stage16(h + (size_t)row * HID + (kc - HID), 1.0f, &As[r][kc]);
    }
  }
  __syncthreads();

  const int wave = threadIdx.x >> 5;
  const int lane = threadIdx.x & 31;
  const int g = lane >> 4;
  const int m = lane & 15;
  const int col = wave * 16 + m;

  v8f c = {};
#pragma unroll
  for (int k0 = 0; k0 < 256; k0 += 32) {
    v16h a = load_a_frag_lds(&As[m][0], k0, g);
    v16h b = load_b_frag(Wt, col, k0, g, 256);
    c = __builtin_amdgcn_wmma_f32_16x16x32_f16(false, a, false, b, (short)0, c,
                                               false, false);
  }

  const float bsum = bl[col] + br[col];
  const float ga = gamma[col], be = beta[col], bmu = mu[col];
  const float rstd = rsqrtf(var[col] + BN_EPS);
  const int rb = blockIdx.x * 16 + g * 8;
#pragma unroll
  for (int r = 0; r < 8; ++r) {
    int orow = rb + r;
    if (orow < n) {
      float v = c[r] + bsum;
      v = ga * (v - bmu) * rstd + be;
      v = fmaxf(v, 0.0f);
      v += h[(size_t)orow * HID + col];  // residual in full f32
      hout[(size_t)orow * HID + col] = v;
    }
  }
}

// ---------------------------------------------------------------------------
// Classifier: out = log_softmax(h @ W_cls + b_cls). One wave32 per node.
// ---------------------------------------------------------------------------
__global__ __launch_bounds__(256) void cls_kernel(
    const float* __restrict__ h, const float* __restrict__ Wc,
    const float* __restrict__ bc, float* __restrict__ out, int n) {
  int t = blockIdx.x * blockDim.x + threadIdx.x;
  int node = t >> 5;
  if (node >= n) return;
  int lane = t & 31;
  const float* hr = h + (size_t)node * HID;
  float z0 = 0.0f, z1 = 0.0f;
#pragma unroll
  for (int i = lane; i < HID; i += 32) {
    float v = hr[i];
    z0 += v * Wc[i * 2 + 0];
    z1 += v * Wc[i * 2 + 1];
  }
#pragma unroll
  for (int off = 16; off > 0; off >>= 1) {
    z0 += __shfl_xor(z0, off, 32);
    z1 += __shfl_xor(z1, off, 32);
  }
  if (lane == 0) {
    z0 += bc[0];
    z1 += bc[1];
    float mx = fmaxf(z0, z1);
    float lse = mx + logf(expf(z0 - mx) + expf(z1 - mx));
    out[(size_t)node * 2 + 0] = z0 - lse;
    out[(size_t)node * 2 + 1] = z1 - lse;
  }
}

// ---------------------------------------------------------------------------
extern "C" void kernel_launch(void* const* d_in, const int* in_sizes, int n_in,
                              void* d_out, int out_size, void* d_ws, size_t ws_size,
                              hipStream_t stream) {
  const float* x  = (const float*)d_in[0];
  const int*   ei = (const int*)d_in[1];
  const float* Wi = (const float*)d_in[2];
  const float* bi = (const float*)d_in[3];
  const float* Wl = (const float*)d_in[4];
  const float* bl = (const float*)d_in[5];
  const float* Wr = (const float*)d_in[6];
  const float* br = (const float*)d_in[7];
  const float* bg = (const float*)d_in[8];
  const float* bb = (const float*)d_in[9];
  const float* bm = (const float*)d_in[10];
  const float* bv = (const float*)d_in[11];
  const float* Wc = (const float*)d_in[12];
  const float* bc = (const float*)d_in[13];
  float* out = (float*)d_out;

  const int n = in_sizes[0] / IN_DIM;
  const int E = in_sizes[1] / 2;
  const int* src = ei;
  const int* dst = ei + E;

  // Workspace: h0 | h1 | agg | cnt | Wti(f16) | Wtl(f16)   (~154 MB @ n=100k)
  float* ha  = (float*)d_ws;
  float* hb  = ha  + (size_t)n * HID;
  float* agg = hb  + (size_t)n * HID;
  float* cnt = agg + (size_t)n * HID;
  _Float16* Wti = (_Float16*)(cnt + n);
  _Float16* Wtl = Wti + (size_t)HID * IN_DIM;

  // weight conversion + degree counts (edges fixed -> once per launch)
  convert_weights_kernel<<<(HID * IN_DIM + N_LAYERS * HID * 256 + 255) / 256,
                           256, 0, stream>>>(Wi, Wl, Wr, Wti, Wtl);
  fill_zero_kernel<<<256, 256, 0, stream>>>(cnt, (size_t)n);
  count_kernel<<<(E + 255) / 256, 256, 0, stream>>>(dst, cnt, E);

  // input projection + ReLU
  in_gemm_kernel<<<(n + 15) / 16, 256, 0, stream>>>(x, Wti, bi, ha, n);

  float* hc = ha;
  float* hn = hb;
  for (int l = 0; l < N_LAYERS; ++l) {
    fill_zero_kernel<<<2048, 256, 0, stream>>>(agg, (size_t)n * HID);
    scatter_kernel<<<(E * 32 + 255) / 256, 256, 0, stream>>>(hc, src, dst, agg, E);
    layer_kernel<<<(n + 15) / 16, 256, 0, stream>>>(
        hc, agg, cnt, Wtl + (size_t)l * HID * 256,
        bl + (size_t)l * HID, br + (size_t)l * HID,
        bg + (size_t)l * HID, bb + (size_t)l * HID,
        bm + (size_t)l * HID, bv + (size_t)l * HID, hn, n);
    float* t = hc; hc = hn; hn = t;
  }

  cls_kernel<<<((size_t)n * 32 + 255) / 256, 256, 0, stream>>>(hc, Wc, bc, out, n);
}
